// MultiHeadAttention_48799418417787
// MI455X (gfx1250) — matrix operand order repaired and checked by reading, not compiled, not run on hardware
//
#include <hip/hip_runtime.h>

// ---------------- MI455X (gfx1250) MHA forward ----------------
// bf16 WMMA everywhere GEMM-shaped; f32 accumulate; flash softmax.
// v3: attention K/V tiles staged per-block into LDS with
//     global_load_async_to_lds_b128 (double-buffered, ASYNCcnt-tracked),
//     WGP-scope prefetch in the GEMM, register-pipelined GEMM K-loop.

typedef __attribute__((ext_vector_type(16))) __bf16 v16bf;
typedef __attribute__((ext_vector_type(8)))  __bf16 v8bf;
typedef __attribute__((ext_vector_type(8)))  float  v8f;

#define BQ   2
#define LSEQ 2048
#define DMODEL 1024
#define NHEAD 16
#define HDIM 64
#define MTOT (BQ*LSEQ)   // 4096

static __device__ __forceinline__ __bf16 f2bf(float f) {
    unsigned u = __builtin_bit_cast(unsigned, f);
    unsigned r = (u + 0x7FFFu + ((u >> 16) & 1u)) >> 16;   // RNE
    unsigned short s = (unsigned short)r;
    return __builtin_bit_cast(__bf16, s);
}
static __device__ __forceinline__ float b2f(__bf16 b) {
    unsigned short s = __builtin_bit_cast(unsigned short, b);
    unsigned u = ((unsigned)s) << 16;
    return __builtin_bit_cast(float, u);
}
static __device__ __forceinline__ v8f v8f_zero() {
    v8f z = {0.f,0.f,0.f,0.f,0.f,0.f,0.f,0.f};
    return z;
}

// A-fragment (16x32 bf16, MxK): lane m=lane&15; lanes<16 hold K 0-7 & 16-23,
// lanes>=16 hold K 8-15 & 24-31 (ISA 7.12.2). base points at tile origin [row0][k0].
static __device__ __forceinline__ v16bf load_a16(const __bf16* base, int ld, int lane) {
    int r   = lane & 15;
    int off = (lane & 16) ? 8 : 0;
    const __bf16* p = base + (size_t)r * ld + off;
    v8bf lo = *(const v8bf*)(p);
    v8bf hi = *(const v8bf*)(p + 16);
    v16bf o;
#pragma unroll
    for (int i = 0; i < 8; ++i) { o[i] = lo[i]; o[i + 8] = hi[i]; }
    return o;
}

// B-fragment (32x16 bf16, KxN) from an N-major source: B[k][n] = src[n][k].
// Lanes 0-15 hold K 0-7 & 16-23, lanes 16-31 hold K 8-15 & 24-31 (matches A-fragment).
static __device__ __forceinline__ v16bf load_b16(const __bf16* nmajor, int ld, int lane) {
    int n    = lane & 15;
    int koff = (lane & 16) ? 8 : 0;
    const __bf16* p = nmajor + (size_t)n * ld + koff;
    v8bf lo = *(const v8bf*)(p);
    v8bf hi = *(const v8bf*)(p + 16);
    v16bf o;
#pragma unroll
    for (int i = 0; i < 8; ++i) { o[i] = lo[i]; o[i + 8] = hi[i]; }
    return o;
}

static __device__ __forceinline__ v8f wmma_bf16(v16bf a, v16bf b, v8f c) {
    return __builtin_amdgcn_wmma_f32_16x16x32_bf16(false, a, false, b, (short)0, c, false, false);
}

// Async global->LDS copy, 16B per lane (cdna5_isa 15.18.3 opcode 98, GV mode).
// lds_off: LDS byte offset VGPR (low 32 bits of generic shared pointer).
static __device__ __forceinline__ void async_copy_b128(unsigned lds_off,
                                                       unsigned long long gaddr) {
    asm volatile("global_load_async_to_lds_b128 %0, %1, off"
                 :: "v"(lds_off), "v"(gaddr) : "memory");
}
static __device__ __forceinline__ void wait_async0() {
    asm volatile("s_wait_asynccnt 0x0" ::: "memory");
}

// ---------------- Stage 1: fp32 -> bf16 convert ----------------
__global__ void cvt_f32_bf16(const float* __restrict__ in, __bf16* __restrict__ out, int n) {
    int i = blockIdx.x * blockDim.x + threadIdx.x;
    if (i < n) out[i] = f2bf(in[i]);
}

// ---------------- Generic bf16 GEMM: C[m][n] = sum_k A[m][k]*W[n][k] ----------------
// 256 threads = 8 waves; block tile 128x128; wave tile 64(M) x 32(N).
// K-loop register double-buffered (two-phase) + WGP-scope prefetch ahead.
__global__ __launch_bounds__(256)
void gemm_bf16_nt(const __bf16* __restrict__ A, const __bf16* __restrict__ W,
                  void* __restrict__ Cout, int M, int N, int K, int c_is_f32) {
    int lane = threadIdx.x & 31;
    int wid  = threadIdx.x >> 5;
    int m0 = blockIdx.x * 128 + (wid & 1) * 64;
    int n0 = blockIdx.y * 128 + (wid >> 1) * 32;

    v8f acc[4][2];
#pragma unroll
    for (int i = 0; i < 4; ++i)
#pragma unroll
        for (int j = 0; j < 2; ++j) acc[i][j] = v8f_zero();

    auto load_set = [&](v16bf* a, v16bf* b, int k0) {
#pragma unroll
        for (int i = 0; i < 4; ++i)
            a[i] = load_a16(A + (size_t)(m0 + 16 * i) * K + k0, K, lane);
#pragma unroll
        for (int j = 0; j < 2; ++j)
            b[j] = load_b16(W + (size_t)(n0 + 16 * j) * K + k0, K, lane);
    };
    auto mma_set = [&](const v16bf* a, const v16bf* b) {
#pragma unroll
        for (int i = 0; i < 4; ++i)
#pragma unroll
            for (int j = 0; j < 2; ++j)
                acc[i][j] = wmma_bf16(a[i], b[j], acc[i][j]);
    };

    v16bf aC[4], bC[2], aN[4], bN[2];
    load_set(aC, bC, 0);

    const int PF = 256;   // prefetch 8 k-steps ahead (one 512B span per row)
    for (int k0 = 0; k0 < K; k0 += 64) {
        load_set(aN, bN, k0 + 32);                // next-phase loads in flight
        if (k0 + PF < K) {
#pragma unroll
            for (int i = 0; i < 4; ++i)
                __builtin_prefetch(A + (size_t)(m0 + 16 * i + (lane & 15)) * K + k0 + PF, 0, 3);
#pragma unroll
            for (int j = 0; j < 2; ++j)
                __builtin_prefetch(W + (size_t)(n0 + 16 * j + (lane & 15)) * K + k0 + PF, 0, 3);
        }
        mma_set(aC, bC);                          // compute overlaps next loads
        if (k0 + 64 < K) load_set(aC, bC, k0 + 64);
        mma_set(aN, bN);
    }

    int nlo = lane & 15;
    int mhi = (lane >> 4) << 3;
    if (c_is_f32) {
        float* C = (float*)Cout;
#pragma unroll
        for (int i = 0; i < 4; ++i)
#pragma unroll
            for (int j = 0; j < 2; ++j)
#pragma unroll
                for (int r = 0; r < 8; ++r)
                    C[(size_t)(m0 + 16 * i + mhi + r) * N + (n0 + 16 * j + nlo)] = acc[i][j][r];
    } else {
        __bf16* C = (__bf16*)Cout;
#pragma unroll
        for (int i = 0; i < 4; ++i)
#pragma unroll
            for (int j = 0; j < 2; ++j)
#pragma unroll
                for (int r = 0; r < 8; ++r)
                    C[(size_t)(m0 + 16 * i + mhi + r) * N + (n0 + 16 * j + nlo)] = f2bf(acc[i][j][r]);
    }
}

// ---------------- Stage 3: RoPE + head reorder ----------------
// q,k,v: bf16 [B,L,H*hd]. Outputs: qr,kr bf16 [B,H,L,hd]; vt bf16 [B,H,hd,L].
// Q gets the 1/sqrt(hd) = 0.125 scale folded in.
__global__ void rope_reorder(const __bf16* __restrict__ q, const __bf16* __restrict__ k,
                             const __bf16* __restrict__ v, __bf16* __restrict__ qr,
                             __bf16* __restrict__ kr, __bf16* __restrict__ vt) {
    int idx = blockIdx.x * blockDim.x + threadIdx.x;     // (b,h,l,d) flat
    int d  = idx & (HDIM - 1);
    int l  = (idx >> 6) & (LSEQ - 1);
    int bh = idx >> 17;                                  // 0..31
    int b  = bh >> 4;
    int h  = bh & 15;

    size_t src  = ((size_t)(b * LSEQ + l)) * DMODEL + h * HDIM + d;
    size_t srcp = ((size_t)(b * LSEQ + l)) * DMODEL + h * HDIM + (d ^ 32);

    float qv = b2f(q[src]),  kv = b2f(k[src]);
    float qp = b2f(q[srcp]), kp = b2f(k[srcp]);

    float freq = __powf(10000.0f, -(float)(d & 31) * (1.0f / 32.0f));
    float ang  = (float)l * freq;
    float c = __cosf(ang), s = __sinf(ang);
    float sgn = (d < 32) ? -1.0f : 1.0f;

    float qo = qv * c + sgn * qp * s;
    float ko = kv * c + sgn * kp * s;

    size_t dst = ((size_t)bh * LSEQ + l) * HDIM + d;
    qr[dst] = f2bf(qo * 0.125f);
    kr[dst] = f2bf(ko);
    vt[((size_t)bh * HDIM + d) * LSEQ + l] = f2bf(b2f(v[src]));
}

// ---------------- Stage 4: flash attention ----------------
// One wave per (b,h, 16 q-rows). 128 threads/block = 4 waves; the 4 waves of a
// block share the same (b,h), so K/V tiles are staged once per block in LDS via
// double-buffered global_load_async_to_lds_b128 (copy overlaps a full iteration).
__global__ __launch_bounds__(128)
void attn_flash(const __bf16* __restrict__ Q, const __bf16* __restrict__ Kc,
                const __bf16* __restrict__ Vt, __bf16* __restrict__ Oa) {
    // per buffer: K tile 32x64 (elems 0..2047), V tile 64x32 (elems 2048..4095)
    __shared__ __align__(128) __bf16 kvbuf[2][4096];
    __shared__ __align__(64)  __bf16 plds[4][16 * 32];   // per-wave P tile (16x32)

    int tid  = threadIdx.x;
    int lane = tid & 31;
    int wid  = tid >> 5;
    int gw   = blockIdx.x * 4 + wid;    // 0..4095
    int bh   = gw >> 7;                 // 0..31 (same for all 4 waves of a block)
    int qt   = gw & 127;                // q tile within (b,h)
    int b    = bh >> 4;
    int h    = bh & 15;

    const __bf16* qbase = Q  + ((size_t)bh * LSEQ + qt * 16) * HDIM;
    const __bf16* kbase = Kc + (size_t)bh * LSEQ * HDIM;
    const __bf16* vbase = Vt + (size_t)bh * HDIM * LSEQ;

    // LDS byte offsets (generic shared pointer keeps LDS offset in addr[31:0])
    unsigned ldsb[2];
    ldsb[0] = (unsigned)(uintptr_t)(void*)&kvbuf[0][0];
    ldsb[1] = (unsigned)(uintptr_t)(void*)&kvbuf[1][0];

    // stage one 32-key K/V tile: 8KB total, 128 threads x 4 async b128 copies
    auto stage = [&](unsigned lb, int kt) {
        const char* ksrc = (const char*)(kbase + (size_t)kt * HDIM);  // contiguous 4KB
#pragma unroll
        for (int i = 0; i < 2; ++i)
            async_copy_b128(lb + tid * 16 + i * 2048,
                            (unsigned long long)(uintptr_t)(ksrc + tid * 16 + i * 2048));
#pragma unroll
        for (int i = 0; i < 2; ++i) {
            int c   = i * 128 + tid;      // 256 chunks of 16B
            int row = c >> 2;             // d row 0..63
            int q4  = c & 3;              // 16B chunk within 64B row
            async_copy_b128(lb + 4096 + row * 64 + q4 * 16,
                            (unsigned long long)(uintptr_t)
                                ((const char*)vbase + ((size_t)row * LSEQ + kt) * 2 + q4 * 16));
        }
    };

    v16bf qa0 = load_a16(qbase + 0,  HDIM, lane);
    v16bf qa1 = load_a16(qbase + 32, HDIM, lane);

    float mrow[8], lrow[8];
    v8f o[4];
#pragma unroll
    for (int j = 0; j < 8; ++j) { mrow[j] = -1e30f; lrow[j] = 0.f; }
#pragma unroll
    for (int g = 0; g < 4; ++g) o[g] = v8f_zero();

    int nn = lane & 15;
    int mh = (lane >> 4) << 3;
    __bf16* pw = &plds[wid][0];

    stage(ldsb[0], 0);                  // prologue: tile 0 in flight

    for (int kt = 0; kt < LSEQ; kt += 32) {
        int bi = (kt >> 5) & 1;
        wait_async0();                  // this wave's copies for tile kt landed
        __syncthreads();                // all waves' shares landed; also proves
                                        // everyone finished reading buf bi^1
        if (kt + 32 < LSEQ) stage(ldsb[bi ^ 1], kt + 32);   // overlap next copy

        const __bf16* kcur = &kvbuf[bi][0];      // 32x64, ld=64
        const __bf16* vcur = &kvbuf[bi][2048];   // 64x32, ld=32

        v16bf kb0 = load_b16(kcur + 0,            64, lane);
        v16bf kb1 = load_b16(kcur + 32,           64, lane);
        v16bf kb2 = load_b16(kcur + 16 * 64,      64, lane);
        v16bf kb3 = load_b16(kcur + 16 * 64 + 32, 64, lane);

        v8f s0 = v8f_zero(), s1 = v8f_zero();
        s0 = wmma_bf16(qa0, kb0, s0);
        s0 = wmma_bf16(qa1, kb1, s0);
        s1 = wmma_bf16(qa0, kb2, s1);
        s1 = wmma_bf16(qa1, kb3, s1);

        v16bf vb[4];
#pragma unroll
        for (int g = 0; g < 4; ++g)
            vb[g] = load_b16(vcur + (g * 16) * 32, 32, lane);

        // online softmax per row (rows live per-VGPR-index within 16-lane half)
#pragma unroll
        for (int j = 0; j < 8; ++j) {
            float tmax = fmaxf(s0[j], s1[j]);
            tmax = fmaxf(tmax, __shfl_xor(tmax, 1));
            tmax = fmaxf(tmax, __shfl_xor(tmax, 2));
            tmax = fmaxf(tmax, __shfl_xor(tmax, 4));
            tmax = fmaxf(tmax, __shfl_xor(tmax, 8));
            float mnew = fmaxf(mrow[j], tmax);
            float al = __expf(mrow[j] - mnew);
            float p0 = __expf(s0[j] - mnew);
            float p1 = __expf(s1[j] - mnew);
            float ps = p0 + p1;
            ps += __shfl_xor(ps, 1);
            ps += __shfl_xor(ps, 2);
            ps += __shfl_xor(ps, 4);
            ps += __shfl_xor(ps, 8);
            lrow[j] = lrow[j] * al + ps;
            mrow[j] = mnew;
            o[0][j] *= al; o[1][j] *= al; o[2][j] *= al; o[3][j] *= al;
            s0[j] = p0; s1[j] = p1;
        }

        // P: C-layout -> LDS -> A-layout fragment (wave-private, LDS is in-order)
#pragma unroll
        for (int j = 0; j < 8; ++j) {
            pw[(mh + j) * 32 + nn]      = f2bf(s0[j]);
            pw[(mh + j) * 32 + 16 + nn] = f2bf(s1[j]);
        }
        v16bf pa = load_a16(pw, 32, lane);

        // O(16x64) += P(16x32) @ V(32x64)
#pragma unroll
        for (int g = 0; g < 4; ++g)
            o[g] = wmma_bf16(pa, vb[g], o[g]);
    }

    // epilogue: normalize and scatter to [B, L, H*hd] (A-operand layout for O-proj)
#pragma unroll
    for (int g = 0; g < 4; ++g)
#pragma unroll
        for (int j = 0; j < 8; ++j) {
            int m = qt * 16 + mh + j;
            int d = g * 16 + nn;
            float val = o[g][j] / lrow[j];
            Oa[((size_t)b * LSEQ + m) * DMODEL + h * HDIM + d] = f2bf(val);
        }
}

// ---------------- host-side launch ----------------
extern "C" void kernel_launch(void* const* d_in, const int* in_sizes, int n_in,
                              void* d_out, int out_size, void* d_ws, size_t ws_size,
                              hipStream_t stream) {
    const float* x  = (const float*)d_in[0];
    const float* wq = (const float*)d_in[1];
    const float* wk = (const float*)d_in[2];
    const float* wv = (const float*)d_in[3];
    const float* wo = (const float*)d_in[4];

    char* w = (char*)d_ws;
    const size_t MB = 1024 * 1024;
    __bf16* x_bf  = (__bf16*)(w + 0);          //  8 MB  [4096,1024]
    __bf16* wq_bf = (__bf16*)(w + 8  * MB);    //  2 MB
    __bf16* wk_bf = (__bf16*)(w + 10 * MB);
    __bf16* wv_bf = (__bf16*)(w + 12 * MB);
    __bf16* wo_bf = (__bf16*)(w + 14 * MB);
    __bf16* qp    = (__bf16*)(w + 16 * MB);    //  8 MB  [B,L,H*hd]
    __bf16* kp    = (__bf16*)(w + 24 * MB);
    __bf16* vp    = (__bf16*)(w + 32 * MB);
    __bf16* qr    = (__bf16*)(w + 40 * MB);    //  8 MB  [B,H,L,hd]
    __bf16* kr    = (__bf16*)(w + 48 * MB);
    __bf16* vt    = (__bf16*)(w + 56 * MB);    //  8 MB  [B,H,hd,L]
    __bf16* ao    = (__bf16*)(w + 64 * MB);    //  8 MB  attention out [B,L,D]

    const int NX = MTOT * DMODEL;      // 4,194,304
    const int NW = DMODEL * DMODEL;    // 1,048,576

    cvt_f32_bf16<<<NX / 256, 256, 0, stream>>>(x,  x_bf,  NX);
    cvt_f32_bf16<<<NW / 256, 256, 0, stream>>>(wq, wq_bf, NW);
    cvt_f32_bf16<<<NW / 256, 256, 0, stream>>>(wk, wk_bf, NW);
    cvt_f32_bf16<<<NW / 256, 256, 0, stream>>>(wv, wv_bf, NW);
    cvt_f32_bf16<<<NW / 256, 256, 0, stream>>>(wo, wo_bf, NW);

    dim3 ggrid(MTOT / 128, DMODEL / 128);      // 32 x 8
    gemm_bf16_nt<<<ggrid, 256, 0, stream>>>(x_bf, wq_bf, qp, MTOT, DMODEL, DMODEL, 0);
    gemm_bf16_nt<<<ggrid, 256, 0, stream>>>(x_bf, wk_bf, kp, MTOT, DMODEL, DMODEL, 0);
    gemm_bf16_nt<<<ggrid, 256, 0, stream>>>(x_bf, wv_bf, vp, MTOT, DMODEL, DMODEL, 0);

    rope_reorder<<<NX / 256, 256, 0, stream>>>(qp, kp, vp, qr, kr, vt);

    attn_flash<<<(BQ * NHEAD * (LSEQ / 16)) / 4, 128, 0, stream>>>(qr, kr, vt, ao);

    gemm_bf16_nt<<<ggrid, 256, 0, stream>>>(ao, wo_bf, d_out, MTOT, DMODEL, DMODEL, 1);
}
